// GCNLayer_HSI_36206574305747
// MI455X (gfx1250) — matrix operand order repaired and checked by reading, not compiled, not run on hardware
//
#include <hip/hip_runtime.h>
#include <hip/hip_bf16.h>
#include <math.h>

#define NMAT 4096
#define DIN  200
#define DTH  256
#define DOUT 128
#define PRESERVE 6712524u   /* int(0.8 * 4096*4097/2) */
#define MASK_THRESH 0.005f

typedef _Float16 half_t;
typedef __attribute__((ext_vector_type(16))) _Float16 v16h;
typedef __attribute__((ext_vector_type(8)))  _Float16 v8h;
typedef __attribute__((ext_vector_type(8)))  float    v8f;

static __device__ __forceinline__ v16h concat8(v8h lo, v8h hi) {
  return __builtin_shufflevector(lo, hi, 0,1,2,3,4,5,6,7,8,9,10,11,12,13,14,15);
}
static __device__ __forceinline__ v8f zero8() {
  v8f z = {0.f,0.f,0.f,0.f,0.f,0.f,0.f,0.f};
  return z;
}

// ---------------------------------------------------------------------------
// BatchNorm: fold (gamma, beta, mu, var) into per-column affine a,b so that
// Hn[r,c] = H[r,c]*a[c] + b[c]
// ---------------------------------------------------------------------------
__global__ void bn_stats_kernel(const float* __restrict__ H,
                                const float* __restrict__ gamma,
                                const float* __restrict__ beta,
                                float* __restrict__ a_coef,
                                float* __restrict__ b_coef) {
  const int c = blockIdx.x;
  const int t = threadIdx.x;
  float s = 0.f, sq = 0.f;
  for (int r = t; r < NMAT; r += 256) {
    float x = H[(size_t)r * DIN + c];
    s += x; sq += x * x;
  }
  __shared__ float rs[256], rq[256];
  rs[t] = s; rq[t] = sq; __syncthreads();
  for (int o = 128; o > 0; o >>= 1) {
    if (t < o) { rs[t] += rs[t + o]; rq[t] += rq[t + o]; }
    __syncthreads();
  }
  if (t == 0) {
    float mu  = rs[0] * (1.f / NMAT);
    float var = rq[0] * (1.f / NMAT) - mu * mu;
    float a   = gamma[c] * rsqrtf(var + 1e-5f);
    a_coef[c] = a;
    b_coef[c] = beta[c] - mu * a;
  }
}

// ---------------------------------------------------------------------------
// Hx = Hn @ W_theta + b_theta, then f = Hx / max(||Hx||,1e-12) stored as f16
// one block (256 thr) per row; D_TH == 256
// ---------------------------------------------------------------------------
__global__ void hx_f_kernel(const float* __restrict__ H,
                            const float* __restrict__ a_coef,
                            const float* __restrict__ b_coef,
                            const float* __restrict__ W_theta,
                            const float* __restrict__ b_theta,
                            half_t* __restrict__ fhalf) {
  const int i = blockIdx.x, t = threadIdx.x;
  __shared__ float hn[DIN];
  __shared__ float red[DTH];
  if (t < DIN) hn[t] = H[(size_t)i * DIN + t] * a_coef[t] + b_coef[t];
  __syncthreads();
  float acc = b_theta[t];
  for (int k = 0; k < DIN; ++k) acc += hn[k] * W_theta[(size_t)k * DTH + t];
  red[t] = acc * acc; __syncthreads();
  for (int o = 128; o > 0; o >>= 1) {
    if (t < o) red[t] += red[t + o];
    __syncthreads();
  }
  float inv = 1.f / fmaxf(sqrtf(red[0]), 1e-12f);
  fhalf[(size_t)i * DTH + t] = (_Float16)(acc * inv);
}

// ---------------------------------------------------------------------------
// Xout = Hn @ W_out + b_out   (one block of 128 thr per row)
// ---------------------------------------------------------------------------
__global__ void xout_kernel(const float* __restrict__ H,
                            const float* __restrict__ a_coef,
                            const float* __restrict__ b_coef,
                            const float* __restrict__ W_out,
                            const float* __restrict__ b_out,
                            float* __restrict__ Xout) {
  const int i = blockIdx.x, t = threadIdx.x;
  __shared__ float hn[DIN];
  for (int k = t; k < DIN; k += 128) hn[k] = H[(size_t)i * DIN + k] * a_coef[k] + b_coef[k];
  __syncthreads();
  float acc = b_out[t];
  for (int k = 0; k < DIN; ++k) acc += hn[k] * W_out[(size_t)k * DOUT + t];
  Xout[(size_t)i * DOUT + t] = acc;
}

// ---------------------------------------------------------------------------
// Exact k-th smallest of upper-triangular drop_scores via 4-pass radix select
// on positive-float bit patterns (order preserving for values in [0,1)).
// ---------------------------------------------------------------------------
__global__ void sel_init_kernel(unsigned* __restrict__ state, unsigned* __restrict__ hist) {
  int t = threadIdx.x;
  hist[t] = 0u;
  if (t == 0) { state[0] = 0u; state[1] = PRESERVE; }
}

__global__ void sel_hist_kernel(const float* __restrict__ scores,
                                const unsigned* __restrict__ state,
                                unsigned* __restrict__ hist, int pass) {
  __shared__ unsigned lh[256];
  const int t = threadIdx.x;
  lh[t] = 0u; __syncthreads();
  const unsigned prefix = state[0];
  const int shift = 24 - 8 * pass;
  const unsigned total = (unsigned)NMAT * (unsigned)NMAT;
  for (unsigned idx = blockIdx.x * 256u + (unsigned)t; idx < total; idx += gridDim.x * 256u) {
    int i = (int)(idx >> 12);
    int j = (int)(idx & (NMAT - 1));
    if (j < i) continue;                       // upper triangular (k=0) only
    unsigned key = __float_as_uint(scores[idx]);
    if (pass == 0 || (key >> (shift + 8)) == prefix)
      atomicAdd(&lh[(key >> shift) & 0xFFu], 1u);
  }
  __syncthreads();
  if (lh[t]) atomicAdd(&hist[t], lh[t]);
}

__global__ void sel_scan_kernel(unsigned* __restrict__ state, unsigned* __restrict__ hist,
                                float* __restrict__ cutoff, int pass) {
  if (threadIdx.x == 0) {
    unsigned kk = state[1];
    unsigned prefix = state[0];
    unsigned cum = 0; unsigned bin = 255;
    for (int b = 0; b < 256; ++b) {
      unsigned c = hist[b];
      if (cum + c >= kk) { bin = (unsigned)b; break; }
      cum += c;
    }
    state[0] = (prefix << 8) | bin;
    state[1] = kk - cum;
    if (pass == 3) *cutoff = __uint_as_float((prefix << 8) | bin);
  }
  __syncthreads();
  hist[threadIdx.x] = 0u;                      // reset for next pass
}

// ---------------------------------------------------------------------------
// S = f @ f^T via v_wmma_f32_16x16x32_f16, fused epilogue:
//   sam = acos(clip(S)); A1 = max(sigmoid(exp(-0.2 sam)), 0.1)
//   Bsym[i,j] = A1[i,j] * keepUpper(min(i,j),max(i,j))      (A1 symmetric)
//   A = Bsym*mask + I; A = (A > 0.7) ? A : 0   -> written straight to d_out
// block = 256 thr (8 waves); block tile 128x128, wave tile 64x32
// ---------------------------------------------------------------------------
__global__ void __launch_bounds__(256) sam_adj_kernel(
    const half_t* __restrict__ f, const float* __restrict__ adj,
    const float* __restrict__ drop, const float* __restrict__ cutoffp,
    float* __restrict__ Aout) {
  const int tid = threadIdx.x;
  const int ln  = tid & 31;
  const int w   = tid >> 5;        // wave 0..7
  const int wm  = w >> 2;          // 0..1
  const int wn  = w & 3;           // 0..3
  const int rowBase = blockIdx.y * 128 + wm * 64;
  const int colBase = blockIdx.x * 128 + wn * 32;
  const int m     = ln & 15;
  const int koffA = (ln >> 4) * 8;   // A-frag: lanes 0-15 K{0..7,16..23}, 16-31 K{8..15,24..31}
  const int koffB = (ln >> 4) * 8;   // B-frag: same K layout as A-frag

  v8f acc[4][2];
#pragma unroll
  for (int a = 0; a < 4; ++a)
#pragma unroll
    for (int b = 0; b < 2; ++b) acc[a][b] = zero8();

  for (int k0 = 0; k0 < DTH; k0 += 32) {
    v16h af[4], bf[2];
#pragma unroll
    for (int tm = 0; tm < 4; ++tm) {
      const half_t* p = f + (size_t)(rowBase + tm * 16 + m) * DTH + k0 + koffA;
      af[tm] = concat8(*(const v8h*)p, *(const v8h*)(p + 16));
    }
#pragma unroll
    for (int tn = 0; tn < 2; ++tn) {
      const half_t* p = f + (size_t)(colBase + tn * 16 + m) * DTH + k0 + koffB;
      bf[tn] = concat8(*(const v8h*)p, *(const v8h*)(p + 16));
    }
#pragma unroll
    for (int tm = 0; tm < 4; ++tm)
#pragma unroll
      for (int tn = 0; tn < 2; ++tn)
        acc[tm][tn] = __builtin_amdgcn_wmma_f32_16x16x32_f16(
            false, af[tm], false, bf[tn], (short)0, acc[tm][tn], false, false);
  }

  const float cutoff = *cutoffp;
  const int mhalf = (ln >> 4) * 8;
#pragma unroll
  for (int tm = 0; tm < 4; ++tm) {
#pragma unroll
    for (int tn = 0; tn < 2; ++tn) {
#pragma unroll
      for (int r = 0; r < 8; ++r) {
        int i = rowBase + tm * 16 + mhalf + r;
        int j = colBase + tn * 16 + m;
        float s = acc[tm][tn][r];
        s = fminf(fmaxf(s, -1.f), 1.f);
        float sam = acosf(s);
        float e   = expf(-0.2f * sam);
        float sg  = 1.f / (1.f + expf(-e));
        float a1  = fmaxf(sg, 0.1f);
        int u = (i < j) ? i : j;
        int v = (i < j) ? j : i;
        float keep = (drop[(size_t)u * NMAT + v] <= cutoff) ? 1.f : 0.f;
        float mk = (fminf(adj[(size_t)i * NMAT + j], adj[(size_t)j * NMAT + i]) < MASK_THRESH)
                       ? 1.f : 0.f;
        float val = a1 * keep * mk;
        if (i == j) val += 1.f;
        val = (val > 0.7f) ? val : 0.f;
        Aout[(size_t)i * NMAT + j] = val;
      }
    }
  }
}

// ---------------------------------------------------------------------------
// d[i] = (sum_j A[i,j])^(-1/2)   (self-loop guarantees sum >= 1)
// ---------------------------------------------------------------------------
__global__ void rowsum_kernel(const float* __restrict__ A, float* __restrict__ d) {
  const int i = blockIdx.x, t = threadIdx.x;
  float s = 0.f;
  for (int j = t; j < NMAT; j += 256) s += A[(size_t)i * NMAT + j];
  __shared__ float rs[256];
  rs[t] = s; __syncthreads();
  for (int o = 128; o > 0; o >>= 1) {
    if (t < o) rs[t] += rs[t + o];
    __syncthreads();
  }
  if (t == 0) d[i] = rsqrtf(rs[0]);
}

// ---------------------------------------------------------------------------
// Yt[t, j] = f16( d[j] * Xout[j, t] )  — transposed so WMMA B-frags load
// two contiguous 16B runs per lane
// ---------------------------------------------------------------------------
__global__ void ytrans_kernel(const float* __restrict__ Xout, const float* __restrict__ d,
                              half_t* __restrict__ Yt) {
  int idx = blockIdx.x * 256 + threadIdx.x;      // 4096*128 elems
  int j = idx >> 7, t = idx & 127;
  Yt[(size_t)t * NMAT + j] = (_Float16)(d[j] * Xout[idx]);
}

// ---------------------------------------------------------------------------
// out = leaky_relu( d_i * (A @ Yt^T) )  via v_wmma_f32_16x16x32_f16
// block = 256 thr (8 waves); wave tile 16 rows x 128 cols, K = 4096
// ---------------------------------------------------------------------------
__global__ void __launch_bounds__(256) outgemm_kernel(
    const float* __restrict__ A, const half_t* __restrict__ Yt,
    const float* __restrict__ d, float* __restrict__ out) {
  const int tid = threadIdx.x;
  const int ln  = tid & 31;
  const int w   = tid >> 5;                 // 0..7
  const int rowBase = blockIdx.x * 128 + w * 16;
  const int m     = ln & 15;
  const int koffA = (ln >> 4) * 8;
  const int koffB = (ln >> 4) * 8;

  v8f acc[8];
#pragma unroll
  for (int tn = 0; tn < 8; ++tn) acc[tn] = zero8();

  for (int k0 = 0; k0 < NMAT; k0 += 32) {
    const float* ap = A + (size_t)(rowBase + m) * NMAT + k0 + koffA;
    v8f x0 = *(const v8f*)ap;
    v8f x1 = *(const v8f*)(ap + 16);
    v16h af;
#pragma unroll
    for (int e = 0; e < 8; ++e) { af[e] = (_Float16)x0[e]; af[8 + e] = (_Float16)x1[e]; }
#pragma unroll
    for (int tn = 0; tn < 8; ++tn) {
      const half_t* bp = Yt + (size_t)(tn * 16 + m) * NMAT + k0 + koffB;
      v16h bf = concat8(*(const v8h*)bp, *(const v8h*)(bp + 16));
      acc[tn] = __builtin_amdgcn_wmma_f32_16x16x32_f16(
          false, af, false, bf, (short)0, acc[tn], false, false);
    }
  }

  const int mhalf = (ln >> 4) * 8;
#pragma unroll
  for (int tn = 0; tn < 8; ++tn) {
#pragma unroll
    for (int r = 0; r < 8; ++r) {
      int i   = rowBase + mhalf + r;
      int col = tn * 16 + m;
      float v = d[i] * acc[tn][r];
      v = (v > 0.f) ? v : 0.01f * v;
      out[(size_t)i * DOUT + col] = v;
    }
  }
}

// ---------------------------------------------------------------------------
extern "C" void kernel_launch(void* const* d_in, const int* in_sizes, int n_in,
                              void* d_out, int out_size, void* d_ws, size_t ws_size,
                              hipStream_t stream) {
  (void)in_sizes; (void)n_in; (void)out_size; (void)ws_size;

  const float* H       = (const float*)d_in[0];
  const float* gamma   = (const float*)d_in[1];
  const float* beta    = (const float*)d_in[2];
  const float* W_theta = (const float*)d_in[3];
  const float* b_theta = (const float*)d_in[4];
  const float* W_out   = (const float*)d_in[5];
  const float* b_out   = (const float*)d_in[6];
  const float* adj     = (const float*)d_in[7];
  const float* drop    = (const float*)d_in[8];

  float* out  = (float*)d_out;                               // [4096,128]
  float* Aout = (float*)d_out + (size_t)NMAT * DOUT;         // [4096,4096]

  char* ws = (char*)d_ws;
  float*    a_coef = (float*)(ws + 0);                       // 200 f
  float*    b_coef = (float*)(ws + 1024);                    // 200 f
  unsigned* state  = (unsigned*)(ws + 2048);                 // {prefix, k}
  float*    cutoff = (float*)(ws + 2048 + 8);
  unsigned* hist   = (unsigned*)(ws + 3072);                 // 256 u32
  float*    dvec   = (float*)(ws + 4096);                    // 4096 f
  float*    Xout   = (float*)(ws + 20480);                   // 4096*128 f  (2 MB)
  half_t*   fhalf  = (half_t*)(ws + 2117632);                // 4096*256 f16 (2 MB)
  half_t*   Yt     = (half_t*)(ws + 4214784);                // 128*4096 f16 (1 MB)

  bn_stats_kernel<<<DIN, 256, 0, stream>>>(H, gamma, beta, a_coef, b_coef);
  hx_f_kernel<<<NMAT, DTH, 0, stream>>>(H, a_coef, b_coef, W_theta, b_theta, fhalf);
  xout_kernel<<<NMAT, DOUT, 0, stream>>>(H, a_coef, b_coef, W_out, b_out, Xout);

  sel_init_kernel<<<1, 256, 0, stream>>>(state, hist);
  for (int p = 0; p < 4; ++p) {
    sel_hist_kernel<<<4096, 256, 0, stream>>>(drop, state, hist, p);
    sel_scan_kernel<<<1, 256, 0, stream>>>(state, hist, cutoff, p);
  }

  dim3 grid_s(NMAT / 128, NMAT / 128);
  sam_adj_kernel<<<grid_s, 256, 0, stream>>>(fhalf, adj, drop, cutoff, Aout);

  rowsum_kernel<<<NMAT, 256, 0, stream>>>(Aout, dvec);
  ytrans_kernel<<<(NMAT * DOUT) / 256, 256, 0, stream>>>(Xout, dvec, Yt);
  outgemm_kernel<<<NMAT / 128, 256, 0, stream>>>(Aout, Yt, dvec, out);
}
